// RLNeighborSelector_58952721105293
// MI455X (gfx1250) — matrix operand order repaired and checked by reading, not compiled, not run on hardware
//
#include <hip/hip_runtime.h>
#include <hip/hip_bf16.h>
#include <math.h>

// ---------------------------------------------------------------------------
// RLNeighborSelector fused edge-MLP for gfx1250 (MI455X), wave32 + WMMA bf16.
//
// Per block: 128 edges. Pipeline entirely in LDS:
//   gather(384 f32 -> bf16)  ->  L1 [384x256]+ReLU  ->  L2 [256x256]
//   -> policy [256x128]+ReLU -> logit dot -> action
//   -> value  [256x128]+ReLU -> z dot     -> BCE term -> block reduce
// GEMMs use v_wmma_f32_16x16x32_bf16. Weights are pre-converted to bf16 in
// d_ws by a prologue kernel (L2-resident); B-fragments cached in VGPRs across
// M-tiles; A-fragment LDS loads batched per M-tile; K-reduction split into
// two independent WMMA accumulate chains for XDL pipe ILP.
// ---------------------------------------------------------------------------

typedef __attribute__((ext_vector_type(16))) __bf16 v16bf;
typedef __attribute__((ext_vector_type(8)))  __bf16 v8bf;
typedef __attribute__((ext_vector_type(4)))  __bf16 v4bf;
typedef __attribute__((ext_vector_type(8)))  float  v8f;

#define M_TILE   128      // edges per block
#define SA       392      // LDS stride (elems) of 384-wide state (16B-aligned rows)
#define SH       264      // LDS stride of 256-wide activations
#define SP       136      // LDS stride of 128-wide activations
#define HDR_B    2048                              // sRow/sCol/sRed header bytes
#define STATE_B  (M_TILE * SA * 2)                 // 100352
#define H_B      (M_TILE * SH * 2)                 // 67584
#define SMEM_B   (HDR_B + STATE_B + H_B)           // 169984 (< 320KB WGP LDS)

// bf16 weight cache layout inside d_ws (element offsets)
#define W1_OFF   0
#define W2_OFF   (384 * 256)
#define PW1_OFF  (W2_OFF + 256 * 256)
#define VW1_OFF  (PW1_OFF + 256 * 128)
#define WTOT     (VW1_OFF + 256 * 128)             // 229376 elements

// ---- one GEMM stage: sOut[M_TILE x N] = act(sIn[M_TILE x K] @ W[KxN] + b) --
// Wave w owns ntiles {w, w+8, ...}; B-fragments for all K-steps are held in
// VGPRs and reused across the 8 M-tiles. A fragments for a whole M-tile are
// loaded as one batch of ds_load_b128s; the WMMA reduction runs as two
// independent accumulate chains (even/odd K-steps) summed in the epilogue.
template <int K, int N, bool RELU, typename WT>
__device__ __forceinline__ void gemm_stage(const __bf16* __restrict__ sIn, int ldin,
                                           const WT* __restrict__ W,
                                           const float* __restrict__ Bias,
                                           __bf16* __restrict__ sOut, int ldout) {
    const int tid  = threadIdx.x;
    const int wave = tid >> 5;
    const int lane = tid & 31;
    const int lg   = lane >> 4;     // lane group (0: lanes 0-15, 1: lanes 16-31)
    const int lm   = lane & 15;

    constexpr int KSTEPS = K / 32;
    constexpr int NT_PER_WAVE = (N / 16) / 8;

    for (int r = 0; r < NT_PER_WAVE; ++r) {
        const int nbase = (wave + 8 * r) * 16;

        // ---- load B fragments (column n=lm, k = kb+8*lg+[0..7], kb+16+8*lg+[0..7]) ----
        v16bf Bf[KSTEPS];
#pragma unroll
        for (int ks = 0; ks < KSTEPS; ++ks) {
            const int k = ks * 32 + 8 * lg;             // first B row held by lane
            const WT* wp = W + (size_t)k * N + nbase + lm;
            v16bf b;
            if constexpr (sizeof(WT) == 2) {            // bf16 weights: strided loads
#pragma unroll
                for (int i = 0; i < 8; ++i) { b[i] = wp[(size_t)i * N]; b[8 + i] = wp[(size_t)(16 + i) * N]; }
            } else {                                    // fp32 fallback: cvt
#pragma unroll
                for (int i = 0; i < 8; ++i) { b[i] = (__bf16)(float)wp[(size_t)i * N]; b[8 + i] = (__bf16)(float)wp[(size_t)(16 + i) * N]; }
            }
            Bf[ks] = b;
        }
        const float bias = Bias[nbase + lm];            // n is fixed per lane

        // ---- sweep M tiles ----
#pragma unroll
        for (int mt = 0; mt < 8; ++mt) {
            const int mbase = mt * 16;

            // batch ALL A-fragment LDS loads for this M-tile (clause, few waits)
            v16bf Af[KSTEPS];
#pragma unroll
            for (int ks = 0; ks < KSTEPS; ++ks) {
                // A 16x32 bf16 layout: lane holds row m=lm, runs
                // k = kb + 8*lg + [0..7]  and  k = kb + 16 + 8*lg + [0..7]
                const __bf16* ap = sIn + (size_t)(mbase + lm) * ldin + ks * 32 + 8 * lg;
                v8bf lo = *(const v8bf*)ap;
                v8bf hi = *(const v8bf*)(ap + 16);
                v16bf a;
#pragma unroll
                for (int i = 0; i < 8; ++i) { a[i] = lo[i]; a[8 + i] = hi[i]; }
                Af[ks] = a;
            }

            // two independent WMMA accumulate chains (even / odd K-steps)
            v8f acc0 = {0.f, 0.f, 0.f, 0.f, 0.f, 0.f, 0.f, 0.f};
            v8f acc1 = {0.f, 0.f, 0.f, 0.f, 0.f, 0.f, 0.f, 0.f};
#pragma unroll
            for (int ks = 0; ks + 1 < KSTEPS; ks += 2) {
                acc0 = __builtin_amdgcn_wmma_f32_16x16x32_bf16(
                        false, Af[ks],     false, Bf[ks],     (short)0, acc0, false, false);
                acc1 = __builtin_amdgcn_wmma_f32_16x16x32_bf16(
                        false, Af[ks + 1], false, Bf[ks + 1], (short)0, acc1, false, false);
            }
            if constexpr (KSTEPS & 1) {
                acc0 = __builtin_amdgcn_wmma_f32_16x16x32_bf16(
                        false, Af[KSTEPS - 1], false, Bf[KSTEPS - 1], (short)0, acc0, false, false);
            }

            // ---- epilogue: merge chains, bias (+ReLU), f32 -> bf16, store ----
            const int ncol = nbase + lm;
#pragma unroll
            for (int j = 0; j < 8; ++j) {
                const int m = mbase + j + 8 * lg;       // C/D layout
                float v = acc0[j] + acc1[j] + bias;
                if (RELU) v = fmaxf(v, 0.f);
                sOut[(size_t)m * ldout + ncol] = (__bf16)v;
            }
        }
    }
}

// 128-wide dot product of a bf16 LDS row against fp32 weights (vectorized)
__device__ __forceinline__ float head_dot(const __bf16* __restrict__ hr,
                                          const float* __restrict__ w, float init) {
    float acc = init;
#pragma unroll
    for (int c = 0; c < 16; ++c) {
        v8bf h = *(const v8bf*)(hr + c * 8);
#pragma unroll
        for (int i = 0; i < 8; ++i) acc += (float)h[i] * w[c * 8 + i];
    }
    return acc;
}

template <typename WT>
__global__ void __launch_bounds__(256)
rlns_main_kernel(
        const float* __restrict__ x,               // [N_NODES,128]
        const int*   __restrict__ edge_index,      // [2,E]
        const float* __restrict__ gfeat,           // [N_NODES,128]
        const int*   __restrict__ labels,          // [N_NODES]
        const WT*    __restrict__ se_w1, const float* __restrict__ se_b1,
        const WT*    __restrict__ se_w2, const float* __restrict__ se_b2,
        const WT*    __restrict__ pw1,   const float* __restrict__ pb1,
        const float* __restrict__ pw2,   const float* __restrict__ pb2,
        const WT*    __restrict__ vw1,   const float* __restrict__ vb1,
        const float* __restrict__ vw2,   const float* __restrict__ vb2,
        float* __restrict__ out,                   // [1 + E]: loss, actions
        int E, float inv_E) {
    extern __shared__ char smem[];
    int*    sRow   = (int*)smem;                       // [128]
    int*    sCol   = sRow + M_TILE;                    // [128]
    float*  sRed   = (float*)(sCol + M_TILE);          // [256]
    __bf16* sState = (__bf16*)(smem + HDR_B);          // [128 x SA] (384 used)
    __bf16* sH     = (__bf16*)(smem + HDR_B + STATE_B);// [128 x SH] (256 used)
    __bf16* sC     = sState;                           // L2 out reuses state
    __bf16* sP     = sH;                               // head hidden reuses sH

    const int tid = threadIdx.x;
    const int e0  = blockIdx.x * M_TILE;

    // ---- edge indices ----
    if (tid < M_TILE) {
        const int ge = e0 + tid;
        if (ge < E) { sRow[tid] = edge_index[ge]; sCol[tid] = edge_index[E + ge]; }
        else        { sRow[tid] = 0;              sCol[tid] = 0; }
    }
    __syncthreads();

    // ---- gather + concat + bf16 convert: state[e] = x[row]|x[col]|g[col] ----
    for (int i = tid; i < M_TILE * 96; i += 256) {     // 96 float4 per row
        const int e = i / 96;
        const int c = (i - e * 96) * 4;
        float4 v = make_float4(0.f, 0.f, 0.f, 0.f);
        if (e0 + e < E) {
            const float* src;
            if (c < 128)      src = x     + (size_t)sRow[e] * 128 + c;
            else if (c < 256) src = x     + (size_t)sCol[e] * 128 + (c - 128);
            else              src = gfeat + (size_t)sCol[e] * 128 + (c - 256);
            v = *(const float4*)src;
        }
        v4bf b; b[0] = (__bf16)v.x; b[1] = (__bf16)v.y; b[2] = (__bf16)v.z; b[3] = (__bf16)v.w;
        *(v4bf*)(sState + (size_t)e * SA + c) = b;
    }
    __syncthreads();

    // ---- state encoder ----
    gemm_stage<384, 256, true,  WT>(sState, SA, se_w1, se_b1, sH, SH);
    __syncthreads();
    gemm_stage<256, 256, false, WT>(sH, SH, se_w2, se_b2, sC, SH);  // h (no ReLU)
    __syncthreads();

    // ---- policy head ----
    gemm_stage<256, 128, true, WT>(sC, SH, pw1, pb1, sP, SP);
    __syncthreads();
    if (tid < M_TILE && (e0 + tid) < E) {
        const float logit = head_dot(sP + (size_t)tid * SP, pw2, pb2[0]);
        out[1 + e0 + tid] = (logit > 0.f) ? 1.0f : 0.0f; // sigmoid(z)>0.5 <=> z>0
    }
    __syncthreads();                                      // before sP is reused

    // ---- value head + BCE-with-logits ----
    gemm_stage<256, 128, true, WT>(sC, SH, vw1, vb1, sP, SP);
    __syncthreads();
    float term = 0.f;
    if (tid < M_TILE && (e0 + tid) < E) {
        const float z = head_dot(sP + (size_t)tid * SP, vw2, vb2[0]);
        const float y = (float)labels[sRow[tid]];
        term = fmaxf(z, 0.f) - z * y + log1pf(expf(-fabsf(z)));
    }
    sRed[tid] = term;
    __syncthreads();
    if (tid == 0) {
        float s = 0.f;
        for (int i = 0; i < 256; ++i) s += sRed[i];
        atomicAdd(out, s * inv_E);
    }
}

// ---- prologue: convert all GEMM weights fp32 -> bf16 into d_ws ----
__global__ void rlns_cvt_weights(const float* __restrict__ w1,
                                 const float* __restrict__ w2,
                                 const float* __restrict__ p1,
                                 const float* __restrict__ v1,
                                 __bf16* __restrict__ dst) {
    for (int i = blockIdx.x * blockDim.x + threadIdx.x; i < WTOT;
         i += gridDim.x * blockDim.x) {
        float v;
        if (i < W2_OFF)       v = w1[i];
        else if (i < PW1_OFF) v = w2[i - W2_OFF];
        else if (i < VW1_OFF) v = p1[i - PW1_OFF];
        else                  v = v1[i - VW1_OFF];
        dst[i] = (__bf16)v;
    }
}

__global__ void rlns_zero_loss(float* out) { out[0] = 0.f; }

extern "C" void kernel_launch(void* const* d_in, const int* in_sizes, int n_in,
                              void* d_out, int out_size, void* d_ws, size_t ws_size,
                              hipStream_t stream) {
    const float* x     = (const float*)d_in[0];
    const int*   ei    = (const int*)  d_in[1];
    const float* gfeat = (const float*)d_in[2];
    const int*   lab   = (const int*)  d_in[3];
    const float* se_w1 = (const float*)d_in[4];
    const float* se_b1 = (const float*)d_in[5];
    const float* se_w2 = (const float*)d_in[6];
    const float* se_b2 = (const float*)d_in[7];
    const float* pw1   = (const float*)d_in[8];
    const float* pb1   = (const float*)d_in[9];
    const float* pw2   = (const float*)d_in[10];
    const float* pb2   = (const float*)d_in[11];
    const float* vw1   = (const float*)d_in[12];
    const float* vb1   = (const float*)d_in[13];
    const float* vw2   = (const float*)d_in[14];
    const float* vb2   = (const float*)d_in[15];
    float* out = (float*)d_out;

    const int E = in_sizes[1] / 2;                 // edge_index is [2, E]
    const int nblocks = (E + M_TILE - 1) / M_TILE;

    rlns_zero_loss<<<1, 1, 0, stream>>>(out);

    if (ws_size >= (size_t)WTOT * 2) {
        // fast path: bf16 weight cache in workspace
        __bf16* wc = (__bf16*)d_ws;
        rlns_cvt_weights<<<224, 256, 0, stream>>>(se_w1, se_w2, pw1, vw1, wc);
        rlns_main_kernel<__bf16><<<nblocks, 256, SMEM_B, stream>>>(
            x, ei, gfeat, lab,
            wc + W1_OFF, se_b1, wc + W2_OFF, se_b2,
            wc + PW1_OFF, pb1, pw2, pb2,
            wc + VW1_OFF, vb1, vw2, vb2,
            out, E, 1.0f / (float)E);
    } else {
        // fallback: fp32 weights converted on the fly
        rlns_main_kernel<float><<<nblocks, 256, SMEM_B, stream>>>(
            x, ei, gfeat, lab,
            se_w1, se_b1, se_w2, se_b2,
            pw1, pb1, pw2, pb2,
            vw1, vb1, vw2, vb2,
            out, E, 1.0f / (float)E);
    }
}
